// KANLayer_18700287607285
// MI455X (gfx1250) — compile-verified, matrix operand order checked by reading, not run on hardware
//
#include <hip/hip_runtime.h>

typedef __attribute__((ext_vector_type(16))) _Float16 v16h;
typedef __attribute__((ext_vector_type(8)))  _Float16 v8h;
typedef __attribute__((ext_vector_type(8)))  float    v8f;
typedef __attribute__((ext_vector_type(4)))  unsigned int uint32x4;
typedef __attribute__((ext_vector_type(8)))  int      int32x8;
typedef __attribute__((ext_vector_type(4)))  int      int32x4;

#define CCH   192                 // channels (Cin == Cout)
#define GG    5
#define KDIM  (CCH * 2 * GG)      // 1920
#define MTOT  (16 * 3136)         // 50176 rows
#define MTILE 128
#define NT    12                  // 12 n-tiles of 16 -> 192
#define KCH   60                  // 60 k-chunks of 32 -> 1920
#define BSTR  40                  // padded LDS stride in halves (80B row: 64B data + 16B pad)
#define BUFBYTES (CCH * BSTR * 2) // 15360 bytes per LDS buffer

#if defined(__has_builtin)
#if __has_builtin(__builtin_amdgcn_tensor_load_to_lds) && __has_builtin(__builtin_amdgcn_s_wait_tensorcnt)
#define HAVE_TDM 1
#endif
#endif

// Tell the compiler "this LDS buffer may have been written by an external agent
// (the TDM engine)". The pointer escape + memory clobber prevents alias analysis
// from folding subsequent ds_loads of a buffer it believes is never stored to.
#define LDS_TOUCH(p) asm volatile("" :: "v"(p) : "memory")

// ---------------------------------------------------------------------------
// Pack coeffs [2, Cout, Cin, G] f32  ->  Wt [Cout][K] f16,  k = trig*960 + g*192 + c
// ---------------------------------------------------------------------------
__global__ __launch_bounds__(256)
void pack_weights(const float* __restrict__ c0, const float* __restrict__ c1,
                  const float* __restrict__ c2, _Float16* __restrict__ wt) {
    const float* src = (blockIdx.y == 0) ? c0 : ((blockIdx.y == 1) ? c1 : c2);
    _Float16* dst = wt + (size_t)blockIdx.y * (CCH * KDIM);
    int idx = blockIdx.x * 256 + threadIdx.x;
    if (idx >= CCH * KDIM) return;
    int o = idx / KDIM;
    int k = idx % KDIM;
    int trig = k / (KDIM / 2);
    int r = k % (KDIM / 2);
    int g = r / CCH;
    int c = r % CCH;
    dst[idx] = (_Float16)src[(((size_t)trig * CCH + o) * CCH + c) * GG + g];
}

#ifdef HAVE_TDM
// Issue a TDM copy of one B chunk: 2D tile, tile_dim0=32 (f16), tile_dim1=192 rows,
// tensor_dim0_stride=1920; LDS padding: +4 dwords every 16 dwords -> 80B row stride.
__device__ __forceinline__ void tdm_stage_chunk(const _Float16* wsrc, unsigned ldsOff,
                                                _Float16* ldsBase) {
    unsigned long long ga = (unsigned long long)(const void*)wsrc;
    uint32x4 g0;
    g0[0] = 1u;                                    // count=1 (valid), user mode
    g0[1] = ldsOff;                                // lds_addr (bytes)
    g0[2] = (unsigned)ga;                          // global_addr[31:0]
    g0[3] = (unsigned)(ga >> 32) | (2u << 30);     // global_addr[56:32] | type=2
    int32x8 g1;
    g1[0] = (1 << 16)      // data_size = 2 bytes
          | (1 << 20)      // pad_enable
          | (3 << 22)      // pad_interval: code 3 -> 16 dwords (64B)
          | (3 << 25);     // pad_amount:   code 3 -> 4 dwords (16B)
    g1[1] = (int)((unsigned)(KDIM & 0xFFFF) << 16);   // tensor_dim0=1920 (low 16 bits)
    g1[2] = (int)((unsigned)(CCH & 0xFFFF) << 16);    // tensor_dim0 hi=0 | tensor_dim1=192 lo
    g1[3] = (int)(32u << 16);                         // tensor_dim1 hi=0 | tile_dim0=32
    g1[4] = CCH;                                      // tile_dim1=192 | tile_dim2=0
    g1[5] = KDIM;                                     // tensor_dim0_stride low32 = 1920
    g1[6] = 0;                                        // stride hi | tensor_dim1_stride lo (unused)
    g1[7] = 0;
    int32x4 gz4; gz4[0] = 0; gz4[1] = 0; gz4[2] = 0; gz4[3] = 0;
    int32x8 gz8;
#pragma unroll
    for (int i = 0; i < 8; ++i) gz8[i] = 0;
    __builtin_amdgcn_tensor_load_to_lds(g0, g1, gz4, gz4, gz8, 0);
    LDS_TOUCH(ldsBase);   // TDM writes blds; make that visible to the optimizer
}
#endif

// ---------------------------------------------------------------------------
// Fused FourierKAN GEMM:  y[m,o] = sum_k F(x)[m,k] * Wt[o][k] + bias[o]
// Double-buffered LDS B staging (TDM if available), register-pipelined B frags.
// ---------------------------------------------------------------------------
__global__ __launch_bounds__(256)
void kan_gemm(const float* __restrict__ xin, const _Float16* __restrict__ wt,
              const float* __restrict__ bias, float* __restrict__ yout) {
    __shared__ _Float16 blds[2][CCH * BSTR];

    const int lane    = threadIdx.x & 31;
    const int wave    = threadIdx.x >> 5;
    const int mbase   = blockIdx.x * MTILE + wave * 16;
    const int halfsel = lane >> 4;           // 0 or 1
    const int mrow    = mbase + (lane & 15); // A-matrix row for this lane
    const int k0      = halfsel * 8;         // this lane's K sub-offset
    const int nrow    = lane & 15;

    auto stage_coop = [&](int kkc, int buf) {
        const _Float16* wsrc = wt + kkc * 32;
        for (int i = threadIdx.x; i < CCH * 4; i += 256) {
            const int n = i >> 2, seg = i & 3;
            *(uint4*)(&blds[buf][n * BSTR + seg * 8]) =
                *(const uint4*)(wsrc + (size_t)n * KDIM + seg * 8);
        }
    };

#ifdef HAVE_TDM
    const bool tdm_leader = (__builtin_amdgcn_readfirstlane(wave) == 0);
    if (tdm_leader) tdm_stage_chunk(wt, 0, &blds[0][0]);
#else
    stage_coop(0, 0);
#endif

    v8f acc[NT];
#pragma unroll
    for (int i = 0; i < NT; ++i)
#pragma unroll
        for (int j = 0; j < 8; ++j) acc[i][j] = 0.0f;

    for (int kk = 0; kk < KCH; ++kk) {
        const int trig = kk / 30;
        const int rem  = kk % 30;
        const int g    = rem / 6;
        const int c0   = (rem % 6) * 32;
        const float kscale = (float)(g + 1);

        // ---- build A fragment in registers: K = k0..k0+7 and k0+16..k0+23 ----
        const float* xr = xin + (size_t)mrow * CCH + c0 + k0;
        const float4 t0 = *(const float4*)(xr);
        const float4 t1 = *(const float4*)(xr + 4);
        const float4 t2 = *(const float4*)(xr + 16);
        const float4 t3 = *(const float4*)(xr + 20);
        float fv[16];
        fv[0]  = t0.x; fv[1]  = t0.y; fv[2]  = t0.z; fv[3]  = t0.w;
        fv[4]  = t1.x; fv[5]  = t1.y; fv[6]  = t1.z; fv[7]  = t1.w;
        fv[8]  = t2.x; fv[9]  = t2.y; fv[10] = t2.z; fv[11] = t2.w;
        fv[12] = t3.x; fv[13] = t3.y; fv[14] = t3.z; fv[15] = t3.w;

        v16h afrag;
        if (trig == 0) {
#pragma unroll
            for (int j = 0; j < 16; ++j) afrag[j] = (_Float16)__cosf(fv[j] * kscale);
        } else {
#pragma unroll
            for (int j = 0; j < 16; ++j) afrag[j] = (_Float16)__sinf(fv[j] * kscale);
        }

        // prefetch next chunk's x row (global_prefetch_b8)
        {
            const int kn   = (kk + 1 < KCH) ? kk + 1 : kk;
            const int c0n  = ((kn % 30) % 6) * 32;
            __builtin_prefetch(xin + (size_t)mrow * CCH + c0n + k0, 0, 1);
        }

        // ---- sync: staging of chunk kk complete, everyone done with other buffer ----
#ifdef HAVE_TDM
        if (tdm_leader) __builtin_amdgcn_s_wait_tensorcnt(0);
#endif
        __syncthreads();
#ifdef HAVE_TDM
        LDS_TOUCH(&blds[0][0]);   // buffer contents published; don't hoist/fold reads
#endif

        // ---- kick off staging of chunk kk+1 (overlaps with WMMA below) ----
        if (kk + 1 < KCH) {
#ifdef HAVE_TDM
            if (tdm_leader)
                tdm_stage_chunk(wt + (size_t)(kk + 1) * 32,
                                ((kk + 1) & 1) ? (unsigned)BUFBYTES : 0u,
                                &blds[0][0]);
#else
            stage_coop(kk + 1, (kk + 1) & 1);
#endif
        }

        // ---- 12 WMMAs, B fragments pipelined one tile ahead ----
        const _Float16* bbase = &blds[kk & 1][0];
        v8h blo[2], bhi[2];
        {
            const _Float16* bp = bbase + nrow * BSTR + k0;
            blo[0] = *(const v8h*)(bp);
            bhi[0] = *(const v8h*)(bp + 16);
        }
#pragma unroll
        for (int nt = 0; nt < NT; ++nt) {
            if (nt + 1 < NT) {
                const _Float16* bp = bbase + ((nt + 1) * 16 + nrow) * BSTR + k0;
                blo[(nt + 1) & 1] = *(const v8h*)(bp);
                bhi[(nt + 1) & 1] = *(const v8h*)(bp + 16);
            }
            v16h bfrag;
#pragma unroll
            for (int j = 0; j < 8; ++j) {
                bfrag[j]     = blo[nt & 1][j];
                bfrag[j + 8] = bhi[nt & 1][j];
            }
            acc[nt] = __builtin_amdgcn_wmma_f32_16x16x32_f16(
                false, afrag, false, bfrag, (short)0, acc[nt], false, false);
        }
    }

    // epilogue: D layout -> lane (L&15)=col, VGPR r -> row = r + halfsel*8
#pragma unroll
    for (int nt = 0; nt < NT; ++nt) {
        const int col = nt * 16 + nrow;
        const float bv = bias[col];
#pragma unroll
        for (int r = 0; r < 8; ++r) {
            const int row = mbase + r + halfsel * 8;
            yout[(size_t)row * CCH + col] = acc[nt][r] + bv;
        }
    }
}

// ---------------------------------------------------------------------------
// Depthwise 3x3 SAME conv on [B,N,C] viewed as [B,C,56,56] + per-channel sum/sumsq
// ---------------------------------------------------------------------------
__global__ __launch_bounds__(192)
void dwconv_stats(const float* __restrict__ y, const float* __restrict__ w,
                  const float* __restrict__ b, float* __restrict__ z,
                  float* __restrict__ sums) {
    const int c = threadIdx.x;
    float wgt[9];
#pragma unroll
    for (int t = 0; t < 9; ++t) wgt[t] = w[c * 9 + t];
    const float bb = b[c];

    float s = 0.0f, sq = 0.0f;
    const int m0 = blockIdx.x * 64;
    for (int i = 0; i < 64; ++i) {
        const int m  = m0 + i;
        const int bi = m / 3136;
        const int r  = m - bi * 3136;
        const int h  = r / 56;
        const int wv = r - h * 56;
        float acc = bb;
#pragma unroll
        for (int dh = -1; dh <= 1; ++dh) {
            const int hh = h + dh;
            if (hh < 0 || hh >= 56) continue;
#pragma unroll
            for (int dw = -1; dw <= 1; ++dw) {
                const int ww = wv + dw;
                if (ww < 0 || ww >= 56) continue;
                acc += y[((size_t)(bi * 3136 + hh * 56 + ww)) * CCH + c] *
                       wgt[(dh + 1) * 3 + (dw + 1)];
            }
        }
        z[(size_t)m * CCH + c] = acc;
        s += acc; sq += acc * acc;
    }
    atomicAdd(&sums[c], s);
    atomicAdd(&sums[CCH + c], sq);
}

__global__ void zero_buf(float* p, int n) {
    int i = blockIdx.x * 256 + threadIdx.x;
    if (i < n) p[i] = 0.0f;
}

// fold BN into scale/shift: out = z*sc + sh, sc = gamma*rstd, sh = beta - mean*sc
__global__ void bn_stats(const float* __restrict__ sums, const float* __restrict__ gamma,
                         const float* __restrict__ beta, float* __restrict__ sc) {
    const int c = threadIdx.x;
    const float inv  = 1.0f / (float)MTOT;
    const float mean = sums[c] * inv;
    const float var  = sums[CCH + c] * inv - mean * mean;
    const float rstd = rsqrtf(var + 1e-5f);
    const float s    = gamma[c] * rstd;
    sc[c]       = s;
    sc[CCH + c] = beta[c] - mean * s;
}

// in-place ReLU(z*sc + sh) — elementwise, safe in place
__global__ __launch_bounds__(192)
void bn_apply(float* __restrict__ z, const float* __restrict__ sc) {
    const int c = threadIdx.x;
    const size_t idx = (size_t)blockIdx.x * CCH + c;
    z[idx] = fmaxf(fmaf(z[idx], sc[c], sc[CCH + c]), 0.0f);
}

// ---------------------------------------------------------------------------
extern "C" void kernel_launch(void* const* d_in, const int* in_sizes, int n_in,
                              void* d_out, int out_size, void* d_ws, size_t ws_size,
                              hipStream_t stream) {
    const float* x = (const float*)d_in[0];
    const float* fc_c[3] = {(const float*)d_in[3], (const float*)d_in[5], (const float*)d_in[7]};
    const float* fc_b[3] = {(const float*)d_in[4], (const float*)d_in[6], (const float*)d_in[8]};
    const float* dw_w[3] = {(const float*)d_in[9],  (const float*)d_in[13], (const float*)d_in[17]};
    const float* dw_b[3] = {(const float*)d_in[10], (const float*)d_in[14], (const float*)d_in[18]};
    const float* bn_g[3] = {(const float*)d_in[11], (const float*)d_in[15], (const float*)d_in[19]};
    const float* bn_b[3] = {(const float*)d_in[12], (const float*)d_in[16], (const float*)d_in[20]};

    char* ws = (char*)d_ws;
    const size_t bufBytes = (size_t)MTOT * CCH * sizeof(float);   // 38.5 MB
    float*    bufA = (float*)ws;
    _Float16* wt   = (_Float16*)(ws + bufBytes);                  // 3 * 192*1920 f16
    float*    sums = (float*)(ws + bufBytes + (size_t)3 * CCH * KDIM * sizeof(_Float16));
    float*    stats = sums + 2 * CCH;
    float*    bufO = (float*)d_out;

    pack_weights<<<dim3((CCH * KDIM + 255) / 256, 3), 256, 0, stream>>>(
        fc_c[0], fc_c[1], fc_c[2], wt);

    const float* cur = x;
    for (int l = 0; l < 3; ++l) {
        float* gout = bufA;               // gemm writes bufA every layer
        float* cout = bufO;               // conv writes d_out every layer (in-place BN after)
        kan_gemm<<<MTOT / MTILE, 256, 0, stream>>>(cur, wt + (size_t)l * CCH * KDIM,
                                                   fc_b[l], gout);
        zero_buf<<<(2 * CCH + 255) / 256, 256, 0, stream>>>(sums, 2 * CCH);
        dwconv_stats<<<MTOT / 64, 192, 0, stream>>>(gout, dw_w[l], dw_b[l], cout, sums);
        bn_stats<<<1, CCH, 0, stream>>>(sums, bn_g[l], bn_b[l], stats);
        bn_apply<<<MTOT, CCH, 0, stream>>>(cout, stats);
        cur = cout;
    }
    (void)in_sizes; (void)n_in; (void)out_size; (void)ws_size;
}
